// PointPillarsModel_45397804319069
// MI455X (gfx1250) — matrix operand order repaired and checked by reading, not compiled, not run on hardware
//
#include <hip/hip_runtime.h>
#include <hip/hip_bf16.h>

typedef __attribute__((ext_vector_type(16))) _Float16 v16h;
typedef __attribute__((ext_vector_type(8)))  float    v8f;
typedef __attribute__((ext_vector_type(4)))  int      v4i;

#define GLOBAL_AS __attribute__((address_space(1)))
#define LDS_AS    __attribute__((address_space(3)))

#define EPSV 1e-5f

#if __has_builtin(__builtin_amdgcn_global_load_async_to_lds_b128)
#define HAVE_ASYNC 1
#else
#define HAVE_ASYNC 0
#endif

#if HAVE_ASYNC
#if __has_builtin(__builtin_amdgcn_s_wait_asynccnt)
#define ASYNC_WAIT() __builtin_amdgcn_s_wait_asynccnt(0)
#else
#define ASYNC_WAIT() asm volatile("s_wait_asynccnt 0x0" ::: "memory")
#endif
__device__ __forceinline__ void async_copy16(const _Float16* src, _Float16* dst) {
    GLOBAL_AS void* gsrc = (GLOBAL_AS void*)src;   // generic -> AS1
    LDS_AS    void* ldst = (LDS_AS void*)dst;      // generic -> AS3
    __builtin_amdgcn_global_load_async_to_lds_b128((GLOBAL_AS v4i*)gsrc,
                                                   (LDS_AS v4i*)ldst, 0, 0);
}
#else
#define ASYNC_WAIT()
__device__ __forceinline__ void async_copy16(const _Float16* src, _Float16* dst) {
    *(uint4*)dst = *(const uint4*)src;
}
#endif

// ---------------------------------------------------------------------------
// PFN: partial per-channel sum/sumsq of x = pillars @ lin_w.T  (lin_b cancels in BN)
// grid (128, 4), block 256. Each blockIdx.y handles a 16-channel chunk.
// ---------------------------------------------------------------------------
__global__ __launch_bounds__(256) void pfn_partial(const float* __restrict__ pillars,
                                                   const float* __restrict__ lin_w,
                                                   float* __restrict__ ps,
                                                   float* __restrict__ psq) {
    __shared__ float wts[16 * 9];
    __shared__ float Ls[8 * 16], Lq[8 * 16];
    const int tid = threadIdx.x;
    const int chunk = blockIdx.y;
    if (tid < 144) wts[tid] = lin_w[chunk * 144 + tid];
    __syncthreads();

    float s[16], q[16];
#pragma unroll
    for (int c = 0; c < 16; ++c) { s[c] = 0.f; q[c] = 0.f; }

    const int ROWS = 20000 * 32;
    for (int row = blockIdx.x * 256 + tid; row < ROWS; row += 128 * 256) {
        float p[9];
        const float* src = pillars + (size_t)row * 9;
#pragma unroll
        for (int d = 0; d < 9; ++d) p[d] = src[d];
#pragma unroll
        for (int c = 0; c < 16; ++c) {
            float v = 0.f;
#pragma unroll
            for (int d = 0; d < 9; ++d) v += wts[c * 9 + d] * p[d];
            s[c] += v; q[c] += v * v;
        }
    }
    const int lane = tid & 31, wv = tid >> 5;
#pragma unroll
    for (int c = 0; c < 16; ++c) {
        float a = s[c], b = q[c];
        for (int off = 16; off; off >>= 1) { a += __shfl_xor(a, off); b += __shfl_xor(b, off); }
        if (lane == 0) { Ls[wv * 16 + c] = a; Lq[wv * 16 + c] = b; }
    }
    __syncthreads();
    if (tid < 16) {
        float a = 0.f, b = 0.f;
        for (int w = 0; w < 8; ++w) { a += Ls[w * 16 + tid]; b += Lq[w * 16 + tid]; }
        ps [blockIdx.x * 64 + chunk * 16 + tid] = a;
        psq[blockIdx.x * 64 + chunk * 16 + tid] = b;
    }
}

__global__ void pfn_stats(const float* __restrict__ ps, const float* __restrict__ psq,
                          const float* __restrict__ g, const float* __restrict__ b,
                          float* __restrict__ scale, float* __restrict__ shift) {
    const int c = threadIdx.x;
    if (c >= 64) return;
    float s = 0.f, q = 0.f;
    for (int i = 0; i < 128; ++i) { s += ps[i * 64 + c]; q += psq[i * 64 + c]; }
    const float cnt = 640000.f;
    float m = s / cnt;
    float v = fmaxf(q / cnt - m * m, 0.f);
    float sc = g[c] * rsqrtf(v + EPSV);
    scale[c] = sc;
    shift[c] = b[c] - m * sc;
}

// one wave per pillar: lane = point index; relu(bn(x)) then max over 32 points
__global__ __launch_bounds__(256) void pfn_feats(const float* __restrict__ pillars,
                                                 const float* __restrict__ lin_w,
                                                 const float* __restrict__ scale,
                                                 const float* __restrict__ shift,
                                                 float* __restrict__ feats) {
    __shared__ float wts[64 * 9];
    const int tid = threadIdx.x;
    for (int i = tid; i < 576; i += 256) wts[i] = lin_w[i];
    __syncthreads();
    const int wv = tid >> 5, lane = tid & 31;
    const int p = blockIdx.x * 8 + wv;
    if (p >= 20000) return;
    float pt[9];
    const float* src = pillars + ((size_t)p * 32 + lane) * 9;
#pragma unroll
    for (int d = 0; d < 9; ++d) pt[d] = src[d];
    for (int c = 0; c < 64; ++c) {
        float v = 0.f;
#pragma unroll
        for (int d = 0; d < 9; ++d) v += wts[c * 9 + d] * pt[d];
        v = fmaxf(v * scale[c] + shift[c], 0.f);
        for (int off = 16; off; off >>= 1) v = fmaxf(v, __shfl_xor(v, off));
        if (lane == 0) feats[(size_t)p * 64 + c] = v;
    }
}

__global__ void zero16(uint4* __restrict__ p, int n) {
    int i = blockIdx.x * 256 + threadIdx.x;
    if (i < n) p[i] = make_uint4(0u, 0u, 0u, 0u);
}

__global__ void scatter_k(const float* __restrict__ feats, const int* __restrict__ coords,
                          _Float16* __restrict__ img) {
    const int p = blockIdx.x, c = threadIdx.x;
    const int y = coords[p * 3 + 1], x = coords[p * 3 + 2];
    if (y >= 0 && y < 666 && x >= 0 && x < 666)
        img[((size_t)(y * 666 + x)) * 64 + c] = (_Float16)feats[(size_t)p * 64 + c];
}

// OIHW f32 -> [K = tap*Cin + cin][Cout] f16
__global__ void wtrans(const float* __restrict__ src, _Float16* __restrict__ dst,
                       int Cin, int Cout, int total) {
    int idx = blockIdx.x * 256 + threadIdx.x;
    if (idx >= total) return;
    int co = idx % Cout;
    int k = idx / Cout;
    int tap = k / Cin, cin = k % Cin;
    dst[idx] = (_Float16)src[(size_t)co * Cin * 9 + cin * 9 + tap];
}

// fused heads: 21 bbox + 3 cls + 8 zero channels -> Cout=32, Cin=256
__global__ void wtrans_head(const float* __restrict__ hbw, const float* __restrict__ hcw,
                            const float* __restrict__ hbb, const float* __restrict__ hcb,
                            _Float16* __restrict__ dst, float* __restrict__ bias) {
    int idx = blockIdx.x * 256 + threadIdx.x;
    if (idx < 32) bias[idx] = (idx < 21) ? hbb[idx] : (idx < 24 ? hcb[idx - 21] : 0.f);
    const int total = 2304 * 32;
    if (idx >= total) return;
    int co = idx % 32;
    int k = idx / 32;
    int tap = k / 256, cin = k % 256;
    float v = 0.f;
    if (co < 21)       v = hbw[(size_t)co * 2304 + cin * 9 + tap];
    else if (co < 24)  v = hcw[(size_t)(co - 21) * 2304 + cin * 9 + tap];
    dst[idx] = (_Float16)v;
}

// ---------------------------------------------------------------------------
// Implicit-GEMM 3x3 SAME conv via v_wmma_f32_16x16x32_f16.
// Block tile: 128 pixels x 32 channels. 8 waves: wave w -> M subtiles
// {2*(w&3), 2*(w&3)+1}, N subtile w>>2. Two WMMAs per K-step sharing one B
// fragment. A tile staged with async global->LDS (ASYNCcnt) when available.
// K ordered tap*Cin+cin; 32 | Cin so every 32-chunk is one tap, contiguous cin.
// mode 0: write y f16 HWC.  mode 1 (heads): write f32 NCHW (+bias) to dout, ch<24.
// ---------------------------------------------------------------------------
__global__ __launch_bounds__(256) void conv_wmma(const _Float16* __restrict__ in,
                                                 const _Float16* __restrict__ w,
                                                 int H, int W, int Cin, int Cout,
                                                 _Float16* __restrict__ yout, int mode,
                                                 const float* __restrict__ bias,
                                                 float* __restrict__ dout) {
    __shared__ __align__(16) _Float16 lA[128 * 40];  // 128 pixels x 32 K (pad 40)
    __shared__ __align__(16) _Float16 lB[32 * 40];   // transposed: [N 0..31][K 0..31]

    const int HW = H * W;
    const int tid = threadIdx.x;
    const int tileM = blockIdx.x;
    const int nBase = blockIdx.y * 32;
    const int Ktot = 9 * Cin;

    // A-tile loader roles: two pixels per thread, one 8-half segment each
    const int am0 = tid >> 2;
    const int am1 = am0 + 64;
    const int aseg = tid & 3;
    const int ap0 = tileM * 128 + am0;
    const int ap1 = tileM * 128 + am1;
    const bool inb0 = (ap0 < HW), inb1 = (ap1 < HW);
    const int ah0 = inb0 ? (ap0 / W) : 0, aw0 = inb0 ? (ap0 % W) : 0;
    const int ah1 = inb1 ? (ap1 / W) : 0, aw1 = inb1 ? (ap1 % W) : 0;

    const int lane = tid & 31;
    const int wv = tid >> 5;
    const int ms = (wv & 3) * 2;     // first of this wave's two 16-row M subtiles
    const int ni = wv >> 2;
    const int half = (lane >> 4) & 1;
    const int lr = lane & 15;

    union { v8f v; float f[8]; } acc0, acc1;
    acc0.v = (v8f){};
    acc1.v = (v8f){};

    int c0 = 0, th = 0, tw = 0;      // incremental K decomposition: no divisions in loop
    for (int kb = 0; kb < Ktot; kb += 32) {
        const int dh = th - 1, dw = tw - 1;
        // ---- stage A tile (async global->LDS; zero-fill padding lanes) ----
        {
            const int hs = ah0 + dh, ws2 = aw0 + dw;
            _Float16* dst = &lA[am0 * 40 + aseg * 8];
            if (inb0 && hs >= 0 && hs < H && ws2 >= 0 && ws2 < W) {
                async_copy16(in + ((size_t)(hs * W + ws2)) * Cin + c0 + aseg * 8, dst);
            } else {
                *(uint4*)dst = make_uint4(0u, 0u, 0u, 0u);
            }
        }
        {
            const int hs = ah1 + dh, ws2 = aw1 + dw;
            _Float16* dst = &lA[am1 * 40 + aseg * 8];
            if (inb1 && hs >= 0 && hs < H && ws2 >= 0 && ws2 < W) {
                async_copy16(in + ((size_t)(hs * W + ws2)) * Cin + c0 + aseg * 8, dst);
            } else {
                *(uint4*)dst = make_uint4(0u, 0u, 0u, 0u);
            }
        }

        // ---- stage B tile (transposed into lB[N][K]) ----
        if (tid < 128) {
            const int kr = tid >> 2;
            const int seg = tid & 3;
            uint4 bv = *(const uint4*)(w + (size_t)(kb + kr) * Cout + nBase + seg * 8);
            _Float16 tmp[8];
            *(uint4*)tmp = bv;
#pragma unroll
            for (int t = 0; t < 8; ++t) lB[(seg * 8 + t) * 40 + kr] = tmp[t];
        }
        if (kb + 32 < Ktot)
            __builtin_prefetch((const void*)(w + (size_t)(kb + 32) * Cout + nBase), 0, 1);

        ASYNC_WAIT();
        __syncthreads();

        // ---- fragments per ISA 16-bit layouts (contiguous 16B LDS reads) ----
        union { v16h v; uint4 u[2]; } af0, af1, bf;
        const _Float16* ar0 = &lA[(ms * 16 + lr) * 40];
        const _Float16* ar1 = ar0 + 16 * 40;
        af0.u[0] = *(const uint4*)(ar0 + half * 8);
        af0.u[1] = *(const uint4*)(ar0 + 16 + half * 8);
        af1.u[0] = *(const uint4*)(ar1 + half * 8);
        af1.u[1] = *(const uint4*)(ar1 + 16 + half * 8);
        const _Float16* br = &lB[(ni * 16 + lr) * 40];
        bf.u[0] = *(const uint4*)(br + half * 8);
        bf.u[1] = *(const uint4*)(br + 16 + half * 8);

        acc0.v = __builtin_amdgcn_wmma_f32_16x16x32_f16(false, af0.v, false, bf.v,
                                                        (short)0, acc0.v, false, false);
        acc1.v = __builtin_amdgcn_wmma_f32_16x16x32_f16(false, af1.v, false, bf.v,
                                                        (short)0, acc1.v, false, false);
        __syncthreads();

        c0 += 32;
        if (c0 == Cin) { c0 = 0; ++tw; if (tw == 3) { tw = 0; ++th; } }
    }

    // ---- epilogue: D layout lane L: N = L&15, VGPR r: M = r + 8*(L>=16) ----
    const int ch = nBase + ni * 16 + lr;
#pragma unroll
    for (int r = 0; r < 8; ++r) {
        const int pA = tileM * 128 + ms * 16 + half * 8 + r;
        const int pB = pA + 16;
        if (mode == 0) {
            if (pA < HW) yout[(size_t)pA * Cout + ch] = (_Float16)acc0.f[r];
            if (pB < HW) yout[(size_t)pB * Cout + ch] = (_Float16)acc1.f[r];
        } else if (ch < 24) {
            if (pA < HW) dout[(size_t)ch * HW + pA] = acc0.f[r] + bias[ch];
            if (pB < HW) dout[(size_t)ch * HW + pB] = acc1.f[r] + bias[ch];
        }
    }
}

// deterministic two-stage BN stats over f16 conv output (HWC)
__global__ __launch_bounds__(256) void bn_partial(const _Float16* __restrict__ y, int E,
                                                  int Cout, int NB,
                                                  float* __restrict__ ps, float* __restrict__ psq) {
    __shared__ float Ls[256], Lq[256];
    const int tid = threadIdx.x, blk = blockIdx.x;
    const int stride = NB * 256;
    float s = 0.f, q = 0.f;
    for (int i = blk * 256 + tid; i < E; i += stride) {
        float v = (float)y[i];
        s += v; q += v * v;
    }
    Ls[tid] = s; Lq[tid] = q;
    __syncthreads();
    if (tid < Cout) {
        const int G = 256 / Cout;
        float ts = 0.f, tq = 0.f;
        for (int g = 0; g < G; ++g) { ts += Ls[tid + g * Cout]; tq += Lq[tid + g * Cout]; }
        const int cc = (blk * 256 + tid) % Cout;   // bijection tid<->cc within block
        ps [blk * Cout + cc] = ts;
        psq[blk * Cout + cc] = tq;
    }
}

__global__ void bn_stats(const float* __restrict__ ps, const float* __restrict__ psq,
                         const float* __restrict__ g, const float* __restrict__ b,
                         int Cout, int NB, float cnt,
                         float* __restrict__ scale, float* __restrict__ shift) {
    const int c = threadIdx.x;
    if (c >= Cout) return;
    float s = 0.f, q = 0.f;
    for (int i = 0; i < NB; ++i) { s += ps[i * Cout + c]; q += psq[i * Cout + c]; }
    float m = s / cnt;
    float v = fmaxf(q / cnt - m * m, 0.f);
    float sc = g[c] * rsqrtf(v + EPSV);
    scale[c] = sc;
    shift[c] = b[c] - m * sc;
}

// fused BN + ReLU (+ optional 2x2 maxpool), f16 HWC -> f16 HWC, 8 channels/thread
__global__ __launch_bounds__(256) void finalize_k(const _Float16* __restrict__ y,
                                                  const float* __restrict__ scale,
                                                  const float* __restrict__ shift,
                                                  int W, int Cout, int Wo, int pool, int Eo8,
                                                  _Float16* __restrict__ out) {
    const int idx = blockIdx.x * 256 + threadIdx.x;
    if (idx >= Eo8) return;
    const int CG = Cout >> 3;
    const int c8 = idx % CG;
    const int op = idx / CG;
    const int wo = op % Wo, ho = op / Wo;
    const int c0 = c8 * 8;
    float sc[8], sh[8];
#pragma unroll
    for (int t = 0; t < 8; ++t) { sc[t] = scale[c0 + t]; sh[t] = shift[c0 + t]; }
    float best[8];
    const int span = pool ? 2 : 1;
    bool first = true;
    for (int dy = 0; dy < span; ++dy)
        for (int dx = 0; dx < span; ++dx) {
            const int h = pool ? (2 * ho + dy) : ho;
            const int w = pool ? (2 * wo + dx) : wo;
            uint4 raw = *(const uint4*)(y + ((size_t)(h * W + w)) * Cout + c0);
            _Float16 tv[8];
            *(uint4*)tv = raw;
#pragma unroll
            for (int t = 0; t < 8; ++t) {
                float u = fmaxf((float)tv[t] * sc[t] + sh[t], 0.f);
                best[t] = first ? u : fmaxf(best[t], u);
            }
            first = false;
        }
    _Float16 ov[8];
#pragma unroll
    for (int t = 0; t < 8; ++t) ov[t] = (_Float16)best[t];
    *(uint4*)(out + ((size_t)(ho * Wo + wo)) * Cout + c0) = *(uint4*)ov;
}

// ---------------------------------------------------------------------------
extern "C" void kernel_launch(void* const* d_in, const int* in_sizes, int n_in,
                              void* d_out, int out_size, void* d_ws, size_t ws_size,
                              hipStream_t stream) {
    const float* pillars = (const float*)d_in[0];
    const int*   coords  = (const int*)d_in[1];
    const float* lin_w   = (const float*)d_in[2];
    const float* pfn_g   = (const float*)d_in[4];
    const float* pfn_b   = (const float*)d_in[5];
    const float* c1_w    = (const float*)d_in[6];
    const float* bn1_g   = (const float*)d_in[8];
    const float* bn1_b   = (const float*)d_in[9];
    const float* c2_w    = (const float*)d_in[10];
    const float* bn2_g   = (const float*)d_in[12];
    const float* bn2_b   = (const float*)d_in[13];
    const float* h1_w    = (const float*)d_in[14];
    const float* hbn_g   = (const float*)d_in[16];
    const float* hbn_b   = (const float*)d_in[17];
    const float* hb_w    = (const float*)d_in[18];
    const float* hb_b    = (const float*)d_in[19];
    const float* hc_w    = (const float*)d_in[20];
    const float* hc_b    = (const float*)d_in[21];
    float* dout = (float*)d_out;

    char* ws = (char*)d_ws;
    size_t off = 0;
    auto alloc = [&](size_t bytes) {
        size_t o = off;
        off = (off + bytes + 255) & ~(size_t)255;
        return o;
    };
    const size_t O_FEATS = alloc((size_t)20000 * 64 * 4);
    const size_t O_IMG   = alloc((size_t)666 * 666 * 64 * 2);
    const size_t O_Y     = alloc((size_t)666 * 666 * 64 * 2);      // conv-out buffer (reused)
    const size_t O_P1    = alloc((size_t)333 * 333 * 64 * 2);
    const size_t O_P2    = alloc((size_t)166 * 166 * 128 * 2);
    const size_t O_H     = alloc((size_t)166 * 166 * 256 * 2);
    const size_t O_W1    = alloc((size_t)576 * 64 * 2);
    const size_t O_W2    = alloc((size_t)576 * 128 * 2);
    const size_t O_WH    = alloc((size_t)1152 * 256 * 2);
    const size_t O_WHD   = alloc((size_t)2304 * 32 * 2);
    const size_t O_HBIAS = alloc(32 * 4);
    const size_t O_PPS   = alloc(128 * 64 * 4);
    const size_t O_PPQ   = alloc(128 * 64 * 4);
    const size_t O_PSC   = alloc(64 * 4);
    const size_t O_PSH   = alloc(64 * 4);
    const size_t O_BPS   = alloc(512 * 256 * 4);
    const size_t O_BPQ   = alloc(512 * 256 * 4);
    const size_t O_BSC   = alloc(256 * 4);
    const size_t O_BSH   = alloc(256 * 4);
    if (off > ws_size) return;   // workspace too small; bail safely

    float*     feats = (float*)(ws + O_FEATS);
    _Float16*  img   = (_Float16*)(ws + O_IMG);
    _Float16*  ybuf  = (_Float16*)(ws + O_Y);
    _Float16*  p1    = (_Float16*)(ws + O_P1);
    _Float16*  p2    = (_Float16*)(ws + O_P2);
    _Float16*  hbuf  = (_Float16*)(ws + O_H);
    _Float16*  w1    = (_Float16*)(ws + O_W1);
    _Float16*  w2    = (_Float16*)(ws + O_W2);
    _Float16*  wh    = (_Float16*)(ws + O_WH);
    _Float16*  whd   = (_Float16*)(ws + O_WHD);
    float*     hbias = (float*)(ws + O_HBIAS);
    float*     pps   = (float*)(ws + O_PPS);
    float*     ppq   = (float*)(ws + O_PPQ);
    float*     psc   = (float*)(ws + O_PSC);
    float*     psh   = (float*)(ws + O_PSH);
    float*     bps   = (float*)(ws + O_BPS);
    float*     bpq   = (float*)(ws + O_BPQ);
    float*     bsc   = (float*)(ws + O_BSC);
    float*     bsh   = (float*)(ws + O_BSH);

    const int NB = 512;

    // --- PFN ---
    pfn_partial<<<dim3(128, 4), 256, 0, stream>>>(pillars, lin_w, pps, ppq);
    pfn_stats<<<1, 64, 0, stream>>>(pps, ppq, pfn_g, pfn_b, psc, psh);
    pfn_feats<<<2500, 256, 0, stream>>>(pillars, lin_w, psc, psh, feats);

    // --- scatter to BEV image (f16 HWC) ---
    {
        int n16 = (666 * 666 * 64 * 2) / 16;
        zero16<<<(n16 + 255) / 256, 256, 0, stream>>>((uint4*)img, n16);
    }
    scatter_k<<<20000, 64, 0, stream>>>(feats, coords, img);

    // --- weight transforms ---
    wtrans<<<(576 * 64 + 255) / 256, 256, 0, stream>>>(c1_w, w1, 64, 64, 576 * 64);
    wtrans<<<(576 * 128 + 255) / 256, 256, 0, stream>>>(c2_w, w2, 64, 128, 576 * 128);
    wtrans<<<(1152 * 256 + 255) / 256, 256, 0, stream>>>(h1_w, wh, 128, 256, 1152 * 256);
    wtrans_head<<<(2304 * 32 + 255) / 256, 256, 0, stream>>>(hb_w, hc_w, hb_b, hc_b, whd, hbias);

    // --- block 1: conv 64->64 over 666x666, BN, ReLU, pool -> 333x333 ---
    conv_wmma<<<dim3((666 * 666 + 127) / 128, 2), 256, 0, stream>>>(img, w1, 666, 666, 64, 64,
                                                                    ybuf, 0, nullptr, nullptr);
    bn_partial<<<NB, 256, 0, stream>>>(ybuf, 666 * 666 * 64, 64, NB, bps, bpq);
    bn_stats<<<1, 256, 0, stream>>>(bps, bpq, bn1_g, bn1_b, 64, NB, (float)(666 * 666), bsc, bsh);
    {
        int Eo8 = 333 * 333 * 64 / 8;
        finalize_k<<<(Eo8 + 255) / 256, 256, 0, stream>>>(ybuf, bsc, bsh, 666, 64, 333, 1, Eo8, p1);
    }

    // --- block 2: conv 64->128 over 333x333, BN, ReLU, pool -> 166x166 ---
    conv_wmma<<<dim3((333 * 333 + 127) / 128, 4), 256, 0, stream>>>(p1, w2, 333, 333, 64, 128,
                                                                    ybuf, 0, nullptr, nullptr);
    bn_partial<<<NB, 256, 0, stream>>>(ybuf, 333 * 333 * 128, 128, NB, bps, bpq);
    bn_stats<<<1, 256, 0, stream>>>(bps, bpq, bn2_g, bn2_b, 128, NB, (float)(333 * 333), bsc, bsh);
    {
        int Eo8 = 166 * 166 * 128 / 8;
        finalize_k<<<(Eo8 + 255) / 256, 256, 0, stream>>>(ybuf, bsc, bsh, 333, 128, 166, 1, Eo8, p2);
    }

    // --- head trunk: conv 128->256 over 166x166, BN, ReLU ---
    conv_wmma<<<dim3((166 * 166 + 127) / 128, 8), 256, 0, stream>>>(p2, wh, 166, 166, 128, 256,
                                                                    ybuf, 0, nullptr, nullptr);
    bn_partial<<<NB, 256, 0, stream>>>(ybuf, 166 * 166 * 256, 256, NB, bps, bpq);
    bn_stats<<<1, 256, 0, stream>>>(bps, bpq, hbn_g, hbn_b, 256, NB, (float)(166 * 166), bsc, bsh);
    {
        int Eo8 = 166 * 166 * 256 / 8;
        finalize_k<<<(Eo8 + 255) / 256, 256, 0, stream>>>(ybuf, bsc, bsh, 166, 256, 166, 0, Eo8, hbuf);
    }

    // --- fused heads: conv 256->32 (21 bbox + 3 cls + pad), f32 NCHW into d_out ---
    conv_wmma<<<dim3((166 * 166 + 127) / 128, 1), 256, 0, stream>>>(hbuf, whd, 166, 166, 256, 32,
                                                                    nullptr, 1, hbias, dout);
}
